// GQASlidingWindowAttention_1846835938082
// MI455X (gfx1250) — matrix operand order repaired and checked by reading, not compiled, not run on hardware
//
#include <hip/hip_runtime.h>
#include <hip/hip_bf16.h>

typedef __bf16 bf16_t;
typedef __attribute__((ext_vector_type(16))) __bf16 v16bf;
typedef __attribute__((ext_vector_type(8)))  __bf16 v8bf;
typedef __attribute__((ext_vector_type(8)))  float  v8f;

#define S_LEN  2048
#define DMODEL 2048
#define HQ_N   16
#define HKV_N  4
#define DH_N   128
#define WIN    512

static __device__ __forceinline__ v16bf pack16(v8bf lo, v8bf hi) {
    return __builtin_shufflevector(lo, hi, 0, 1, 2, 3, 4, 5, 6, 7,
                                           8, 9, 10, 11, 12, 13, 14, 15);
}

// ---------------------------------------------------------------------------
// fp32 -> bf16 conversion (grid-stride)
// ---------------------------------------------------------------------------
__global__ void cvt_f32_bf16(const float* __restrict__ src, bf16_t* __restrict__ dst, int n) {
    int i = blockIdx.x * blockDim.x + threadIdx.x;
    int stride = gridDim.x * blockDim.x;
    for (; i < n; i += stride) dst[i] = (bf16_t)src[i];
}

// fp32 [K,N] -> bf16 transposed [N,K] (coalesced reads)
__global__ void cvt_transpose_bf16(const float* __restrict__ src, bf16_t* __restrict__ dst,
                                   int K, int N) {
    int i = blockIdx.x * blockDim.x + threadIdx.x;
    int stride = gridDim.x * blockDim.x;
    const int total = K * N;
    for (; i < total; i += stride) {
        const int k = i / N;
        const int n = i - k * N;
        dst[(size_t)n * K + k] = (bf16_t)src[i];
    }
}

// ---------------------------------------------------------------------------
// bf16 GEMM: C[M,N] (f32) = A[M,K] * Bt^T, A row-major, Bt is B transposed [N,K].
// 64x64 tile / 256-thread block (8 waves); wave computes 16x32 strip (2 WMMAs).
// A tile staged to LDS via global_load_async_to_lds_b128 (ASYNCcnt path);
// B fragments read directly from Bt as two contiguous b128 loads per lane.
// ---------------------------------------------------------------------------
__global__ __launch_bounds__(256)
void gemm_bf16_wmma(const bf16_t* __restrict__ A, const bf16_t* __restrict__ Bt,
                    float* __restrict__ C, int M, int N, int K) {
    __shared__ bf16_t As[64 * 40];   // 64 rows x 32 cols, padded stride 40 (80B rows)
    const int tid  = threadIdx.x;
    const int lane = tid & 31;
    const int wave = tid >> 5;
    const int m0 = blockIdx.y * 64;
    const int n0 = blockIdx.x * 64;
    const int wm = (wave >> 1) * 16;   // 0,16,32,48
    const int wn = (wave & 1) * 32;    // 0,32

    v8f acc0 = {};
    v8f acc1 = {};

    // cooperative A-tile load: each thread moves 16B (8 bf16)
    const int arow = tid >> 2, acol = (tid & 3) * 8;
    const unsigned as_off = (unsigned)(size_t)(&As[arow * 40 + acol]); // low 32b = LDS addr

    // per-lane fragment geometry
    const int fr = wm + (lane & 15);      // A row
    const int h0 = (lane >> 4) * 8;       // A K-half base
    const int kl = (lane >> 4) * 8;       // B K base
    const bf16_t* bcol0 = Bt + (size_t)(n0 + wn + (lane & 15)) * K;
    const bf16_t* bcol1 = bcol0 + (size_t)16 * K;

    for (int k = 0; k < K; k += 32) {
        __syncthreads();
        {
            const bf16_t* ga = &A[(size_t)(m0 + arow) * K + k + acol];
            asm volatile("global_load_async_to_lds_b128 %0, %1, off"
                         :: "v"(as_off), "v"(ga) : "memory");
            asm volatile("s_wait_asynccnt 0x0" ::: "memory");
        }
        __syncthreads();

        v8bf alo = *(const v8bf*)(&As[fr * 40 + h0]);
        v8bf ahi = *(const v8bf*)(&As[fr * 40 + h0 + 16]);
        const v16bf af = pack16(alo, ahi);

        v8bf b0l = *(const v8bf*)(bcol0 + k + kl);
        v8bf b0h = *(const v8bf*)(bcol0 + k + kl + 16);
        v8bf b1l = *(const v8bf*)(bcol1 + k + kl);
        v8bf b1h = *(const v8bf*)(bcol1 + k + kl + 16);

        acc0 = __builtin_amdgcn_wmma_f32_16x16x32_bf16(false, af, false, pack16(b0l, b0h),
                                                       (short)0, acc0, false, false);
        acc1 = __builtin_amdgcn_wmma_f32_16x16x32_bf16(false, af, false, pack16(b1l, b1h),
                                                       (short)0, acc1, false, false);
    }

    // C/D layout: lane = (m>=8)*16 + n, vgpr r -> m = r + (lane>=16)*8
    const int cn = lane & 15;
    const int mb = (lane >> 4) * 8;
    for (int r = 0; r < 8; ++r) {
        C[(size_t)(m0 + wm + mb + r) * N + n0 + wn + cn]      = acc0[r];
        C[(size_t)(m0 + wm + mb + r) * N + n0 + wn + 16 + cn] = acc1[r];
    }
}

// ---------------------------------------------------------------------------
// RoPE for Q: reads fp32 Q [S, HQ*DH], writes bf16 Q (post-RoPE)
// ---------------------------------------------------------------------------
__global__ void rope_q_kernel(const float* __restrict__ Qf, const float* __restrict__ cosc,
                              const float* __restrict__ sinc, bf16_t* __restrict__ Qb) {
    int i = blockIdx.x * blockDim.x + threadIdx.x;   // over S*HQ*64
    if (i >= S_LEN * HQ_N * 64) return;
    const int d = i & 63;
    const int h = (i >> 6) & (HQ_N - 1);
    const int s = i >> 10;
    const float c  = cosc[s * DH_N + d];
    const float sn = sinc[s * DH_N + d];
    const size_t base = (size_t)s * (HQ_N * DH_N) + h * DH_N + d;
    const float u1 = Qf[base];
    const float u2 = Qf[base + 64];
    Qb[base]      = (bf16_t)(u1 * c - u2 * sn);
    Qb[base + 64] = (bf16_t)(u2 * c + u1 * sn);
}

// RoPE for K: writes bf16 K [S, HKV*DH] and fp32 K [HKV, S, DH] (output cache)
__global__ void rope_k_kernel(const float* __restrict__ Kf, const float* __restrict__ cosc,
                              const float* __restrict__ sinc, bf16_t* __restrict__ Kb,
                              float* __restrict__ Kout) {
    int i = blockIdx.x * blockDim.x + threadIdx.x;   // over S*HKV*64
    if (i >= S_LEN * HKV_N * 64) return;
    const int d = i & 63;
    const int h = (i >> 6) & (HKV_N - 1);
    const int s = i >> 8;
    const float c  = cosc[s * DH_N + d];
    const float sn = sinc[s * DH_N + d];
    const size_t base = (size_t)s * (HKV_N * DH_N) + h * DH_N + d;
    const float u1 = Kf[base];
    const float u2 = Kf[base + 64];
    const float k1 = u1 * c - u2 * sn;
    const float k2 = u2 * c + u1 * sn;
    Kb[base]      = (bf16_t)k1;
    Kb[base + 64] = (bf16_t)k2;
    const size_t ob = (size_t)h * S_LEN * DH_N + (size_t)s * DH_N + d;
    Kout[ob]      = k1;
    Kout[ob + 64] = k2;
}

// V: fp32 [S,HKV*DH] -> fp32 [HKV,S,DH] output cache + bf16 V^T [HKV,DH,S]
__global__ void copy_v_kernel(const float* __restrict__ Vf, bf16_t* __restrict__ Vt,
                              float* __restrict__ Vout) {
    int i = blockIdx.x * blockDim.x + threadIdx.x;   // over S*HKV*DH
    if (i >= S_LEN * HKV_N * DH_N) return;
    const int d = i & (DH_N - 1);
    const int h = (i >> 7) & (HKV_N - 1);
    const int s = i >> 9;
    const float v = Vf[i];
    Vt[((size_t)h * DH_N + d) * S_LEN + s] = (bf16_t)v;
    Vout[(size_t)h * S_LEN * DH_N + (size_t)s * DH_N + d] = v;
}

// ---------------------------------------------------------------------------
// Flash attention with sliding window + GQA. One wave per 16-row Q tile,
// 32 keys per step. All operand fragments are contiguous b128 loads.
// blockIdx.x = head (16), blockIdx.y = q-block (64 rows, 4 waves).
// ---------------------------------------------------------------------------
__global__ __launch_bounds__(128)
void attn_kernel(const bf16_t* __restrict__ Qb, const bf16_t* __restrict__ Kb,
                 const bf16_t* __restrict__ Vt, bf16_t* __restrict__ Ob) {
    __shared__ bf16_t Pl[4 * 16 * 32];
    const int lane = threadIdx.x & 31;
    const int wave = threadIdx.x >> 5;
    const int h  = blockIdx.x;
    const int q0 = blockIdx.y * 64 + wave * 16;
    const int kv = h >> 2;                     // G = HQ/HKV = 4
    bf16_t* P = &Pl[wave * 16 * 32];

    const float scale = 0.08838834764831845f;  // 1/sqrt(128)
    const float NEG   = -3.0e38f;

    // Preload Q A-fragments for all 4 K-chunks of DH=128
    v16bf qf[4];
    {
        const int row = q0 + (lane & 15);
        const int h0  = (lane >> 4) * 8;
        const bf16_t* qp = Qb + (size_t)row * DMODEL + h * DH_N;
        for (int f = 0; f < 4; ++f) {
            v8bf lo = *(const v8bf*)(qp + f * 32 + h0);
            v8bf hi = *(const v8bf*)(qp + f * 32 + h0 + 16);
            qf[f] = pack16(lo, hi);
        }
    }

    v8f vzero = {};
    v8f o[8];
    for (int c = 0; c < 8; ++c) o[c] = vzero;
    float mrow[8], lrow[8];
    for (int r = 0; r < 8; ++r) { mrow[r] = NEG; lrow[r] = 0.f; }

    const int mb = (lane >> 4) * 8;   // row base within C layout
    const int cn = lane & 15;         // column within 16

    int jstart = q0 - (WIN - 1);
    if (jstart < 0) jstart = 0;
    jstart &= ~31;
    const int jend = q0 + 16;

    const int kl = (lane >> 4) * 8;   // B-fragment K base
    const bf16_t* vbase = Vt + (size_t)kv * DH_N * S_LEN;

    for (int jt = jstart; jt < jend; jt += 32) {
        // ---- scores: two 16-key tiles, DH=128 as 4 x K=32 WMMAs each
        v8f s0 = vzero, s1 = vzero;
        {
            const bf16_t* kp0 = Kb + (size_t)(jt + cn) * (HKV_N * DH_N) + kv * DH_N + kl;
            const bf16_t* kp1 = kp0 + (size_t)16 * (HKV_N * DH_N);
            for (int f = 0; f < 4; ++f) {
                v8bf x0 = *(const v8bf*)(kp0 + f * 32);
                v8bf x1 = *(const v8bf*)(kp0 + f * 32 + 16);
                v8bf y0 = *(const v8bf*)(kp1 + f * 32);
                v8bf y1 = *(const v8bf*)(kp1 + f * 32 + 16);
                s0 = __builtin_amdgcn_wmma_f32_16x16x32_bf16(false, qf[f], false, pack16(x0, x1),
                                                             (short)0, s0, false, false);
                s1 = __builtin_amdgcn_wmma_f32_16x16x32_bf16(false, qf[f], false, pack16(y0, y1),
                                                             (short)0, s1, false, false);
            }
        }

        // ---- mask + scale + online softmax (row reductions across 16-lane halves)
        float p0[8], p1[8], corr[8];
        for (int r = 0; r < 8; ++r) {
            const int i  = q0 + mb + r;
            const int j0 = jt + cn;
            const int j1 = jt + 16 + cn;
            float v0 = (j0 <= i && (i - j0) < WIN) ? s0[r] * scale : NEG;
            float v1 = (j1 <= i && (i - j1) < WIN) ? s1[r] * scale : NEG;
            float tmax = fmaxf(v0, v1);
            for (int mk = 1; mk < 16; mk <<= 1)
                tmax = fmaxf(tmax, __shfl_xor(tmax, mk, 32));
            const float mnew = fmaxf(mrow[r], tmax);
            const float e0 = __expf(v0 - mnew);
            const float e1 = __expf(v1 - mnew);
            float rsum = e0 + e1;
            for (int mk = 1; mk < 16; mk <<= 1)
                rsum += __shfl_xor(rsum, mk, 32);
            const float cr = __expf(mrow[r] - mnew);
            lrow[r] = lrow[r] * cr + rsum;
            mrow[r] = mnew;
            corr[r] = cr;
            p0[r] = e0; p1[r] = e1;
        }

        // ---- stage P (C layout -> LDS) and rescale O
        for (int r = 0; r < 8; ++r) {
            P[(mb + r) * 32 + cn]      = (bf16_t)p0[r];
            P[(mb + r) * 32 + 16 + cn] = (bf16_t)p1[r];
            for (int c = 0; c < 8; ++c) o[c][r] *= corr[r];
        }
        asm volatile("s_wait_dscnt 0x0" ::: "memory");  // wave-local LDS transpose fence

        // ---- P as A fragment
        v16bf pf;
        {
            const int row = lane & 15;
            const int h0  = (lane >> 4) * 8;
            v8bf lo = *(const v8bf*)(&P[row * 32 + h0]);
            v8bf hi = *(const v8bf*)(&P[row * 32 + h0 + 16]);
            pf = pack16(lo, hi);
        }

        // ---- O += P * V over 8 DH column tiles (K = 32 keys); V^T gives
        //      contiguous key-runs per lane: two b128 loads per fragment.
        for (int c = 0; c < 8; ++c) {
            const bf16_t* vp = vbase + (size_t)(c * 16 + cn) * S_LEN + jt + kl;
            v8bf vl = *(const v8bf*)(vp);
            v8bf vh = *(const v8bf*)(vp + 16);
            o[c] = __builtin_amdgcn_wmma_f32_16x16x32_bf16(false, pf, false, pack16(vl, vh),
                                                           (short)0, o[c], false, false);
        }
    }

    // ---- normalize and store attn output (bf16, [S, HQ*DH])
    for (int r = 0; r < 8; ++r) {
        const float inv = 1.0f / lrow[r];
        bf16_t* op = Ob + (size_t)(q0 + mb + r) * DMODEL + h * DH_N;
        for (int c = 0; c < 8; ++c)
            op[c * 16 + cn] = (bf16_t)(o[c][r] * inv);
    }
}

// ---------------------------------------------------------------------------
extern "C" void kernel_launch(void* const* d_in, const int* in_sizes, int n_in,
                              void* d_out, int out_size, void* d_ws, size_t ws_size,
                              hipStream_t stream) {
    (void)in_sizes; (void)n_in; (void)out_size; (void)ws_size;
    const float* x    = (const float*)d_in[0];
    const float* cosc = (const float*)d_in[1];
    const float* sinc = (const float*)d_in[2];
    // d_in[3] positions (int64) and d_in[4] attn_mask (bool): realized analytically
    const float* Wq = (const float*)d_in[5];
    const float* Wk = (const float*)d_in[6];
    const float* Wv = (const float*)d_in[7];
    const float* Wo = (const float*)d_in[8];
    float* out  = (float*)d_out;
    float* Kout = out + (size_t)S_LEN * DMODEL;                          // [HKV,S,DH]
    float* Vout = Kout + (size_t)HKV_N * S_LEN * DH_N;                   // [HKV,S,DH]

    char* ws = (char*)d_ws;
    bf16_t* xb  = (bf16_t*)(ws + (0  << 20));   // 8 MB  x bf16 (row-major)
    bf16_t* Wqt = (bf16_t*)(ws + (8  << 20));   // 8 MB  Wq^T bf16 (later reused for Wo^T)
    bf16_t* Wkt = (bf16_t*)(ws + (16 << 20));   // 2 MB  Wk^T
    bf16_t* Wvt = (bf16_t*)(ws + (18 << 20));   // 2 MB  Wv^T
    float*  Qf  = (float*) (ws + (20 << 20));   // 16 MB fp32 Q (pre-RoPE)
    float*  Kf  = (float*) (ws + (36 << 20));   // 4 MB
    float*  Vf  = (float*) (ws + (40 << 20));   // 4 MB
    bf16_t* Qb  = (bf16_t*)(ws + (44 << 20));   // 8 MB  post-RoPE Q bf16
    bf16_t* Kb  = (bf16_t*)(ws + (52 << 20));   // 2 MB  post-RoPE K bf16 [S,HKV*DH]
    bf16_t* Vt  = (bf16_t*)(ws + (54 << 20));   // 2 MB  V^T bf16 [HKV,DH,S]  (56 MB total)
    bf16_t* Ab  = (bf16_t*)(ws + (20 << 20));   // attn out bf16, aliases Qf (free by then)
    bf16_t* Wot = Wqt;                          // reuse after Q GEMM consumed Wq^T

    const int CVT_T = 256, CVT_B = 2048;
    cvt_f32_bf16<<<CVT_B, CVT_T, 0, stream>>>(x, xb, S_LEN * DMODEL);
    cvt_transpose_bf16<<<CVT_B, CVT_T, 0, stream>>>(Wq, Wqt, DMODEL, HQ_N * DH_N);
    cvt_transpose_bf16<<<CVT_B, CVT_T, 0, stream>>>(Wk, Wkt, DMODEL, HKV_N * DH_N);
    cvt_transpose_bf16<<<CVT_B, CVT_T, 0, stream>>>(Wv, Wvt, DMODEL, HKV_N * DH_N);

    // Projections
    gemm_bf16_wmma<<<dim3(32, 32), 256, 0, stream>>>(xb, Wqt, Qf, S_LEN, HQ_N * DH_N, DMODEL);
    gemm_bf16_wmma<<<dim3(8, 32),  256, 0, stream>>>(xb, Wkt, Kf, S_LEN, HKV_N * DH_N, DMODEL);
    gemm_bf16_wmma<<<dim3(8, 32),  256, 0, stream>>>(xb, Wvt, Vf, S_LEN, HKV_N * DH_N, DMODEL);

    // RoPE + KV cache outputs
    rope_q_kernel<<<(S_LEN * HQ_N * 64 + 255) / 256, 256, 0, stream>>>(Qf, cosc, sinc, Qb);
    rope_k_kernel<<<(S_LEN * HKV_N * 64 + 255) / 256, 256, 0, stream>>>(Kf, cosc, sinc, Kb, Kout);
    copy_v_kernel<<<(S_LEN * HKV_N * DH_N + 255) / 256, 256, 0, stream>>>(Vf, Vt, Vout);

    // Attention (16 heads x 32 q-blocks of 64 rows; 4 waves/block)
    attn_kernel<<<dim3(HQ_N, S_LEN / 64), 128, 0, stream>>>(Qb, Kb, Vt, Ab);

    // Output projection (convert Wo^T now that the Wq^T slot is free)
    cvt_transpose_bf16<<<CVT_B, CVT_T, 0, stream>>>(Wo, Wot, HQ_N * DH_N, DMODEL);
    gemm_bf16_wmma<<<dim3(32, 32), 256, 0, stream>>>(Ab, Wot, out, S_LEN, DMODEL, DMODEL);
}
